// MultiHeadAttention_16149077033087
// MI455X (gfx1250) — matrix operand order repaired and checked by reading, not compiled, not run on hardware
//
#include <hip/hip_runtime.h>

// ---------------------------------------------------------------------------
// MHA for MI455X (gfx1250): all matmuls on v_wmma_f32_16x16x32_bf16, wave32.
// Pipeline: 3x projection GEMM (fp32->bf16) -> flash attention -> out GEMM.
// Intermediates live in d_ws as bf16 (32 MB total) and stay resident in L2.
// CDNA5 paths: v_wmma bf16, global_load_async_to_lds_b128 (+ s_wait_asynccnt),
// s_wait_dscnt for the wave-private P bounce, global_prefetch hints.
// The 1/sqrt(Dh) scale is folded into the Q projection; attention processes
// 64-key chunks so the (log2(16)-step) row-max shuffle cost amortizes.
// ---------------------------------------------------------------------------

#define EMBED 1024
#define NHEAD 16
#define HDIM  64
#define TSEQ  2048
#define BATCH 2
#define MROWS (BATCH * TSEQ)   // 4096

typedef __attribute__((ext_vector_type(8)))  float  v8f;
typedef __attribute__((ext_vector_type(16))) __bf16 v16bf;
typedef __attribute__((ext_vector_type(8)))  __bf16 v8bf;

static __device__ __forceinline__ v16bf cat8(v8bf lo, v8bf hi) {
  return __builtin_shufflevector(lo, hi, 0,1,2,3,4,5,6,7,8,9,10,11,12,13,14,15);
}
static __device__ __forceinline__ v8f wmma_bf16(v16bf a, v16bf b, v8f c) {
  return __builtin_amdgcn_wmma_f32_16x16x32_bf16(false, a, false, b, (short)0, c,
                                                 false, false);
}
static __device__ __forceinline__ v8f vzero8() {
  v8f z;
#pragma unroll
  for (int i = 0; i < 8; ++i) z[i] = 0.0f;
  return z;
}
// memory -> LDS async copies (B128 per issue; the instruction offset is
// applied to BOTH the LDS and the global address, so one register pair
// covers a contiguous run).
static __device__ __forceinline__ void async_copy_32B(const void* gaddr,
                                                      void* lds_ptr) {
  unsigned lds_off = (unsigned)(size_t)lds_ptr;  // low 32 bits = LDS offset
  asm volatile(
      "global_load_async_to_lds_b128 %0, %1, off\n\t"
      "global_load_async_to_lds_b128 %0, %1, off offset:16"
      :
      : "v"(lds_off), "v"(gaddr)
      : "memory");
}
static __device__ __forceinline__ void async_copy_64B(const void* gaddr,
                                                      void* lds_ptr) {
  unsigned lds_off = (unsigned)(size_t)lds_ptr;
  asm volatile(
      "global_load_async_to_lds_b128 %0, %1, off\n\t"
      "global_load_async_to_lds_b128 %0, %1, off offset:16\n\t"
      "global_load_async_to_lds_b128 %0, %1, off offset:32\n\t"
      "global_load_async_to_lds_b128 %0, %1, off offset:48"
      :
      : "v"(lds_off), "v"(gaddr)
      : "memory");
}
static __device__ __forceinline__ void wait_async0() {
  asm volatile("s_wait_asynccnt 0x0" ::: "memory");
}

// ---------------------------------------------------------------------------
// GEMM: Y[M,N] = (X[M,K] @ W[N,K]^T + bias) * out_scale   (torch Linear)
// Block tile 128(M) x 64(N), 8 waves as 4x2, each wave 32x32 = 2x2 WMMA tiles.
// XF32:  X is fp32 (convert to bf16 into LDS) else X already bf16 (async copy).
// SPLITH: epilogue writes bf16 in [B,H,T,Dh] layout (split heads).
// OUTF32: epilogue writes fp32 row-major [M,N] (final output), else bf16 [M,N].
// ---------------------------------------------------------------------------
template <bool XF32, bool SPLITH, bool OUTF32>
__global__ __launch_bounds__(256) void gemm_bias(
    const void* __restrict__ Xv, const float* __restrict__ W,
    const float* __restrict__ bias, void* __restrict__ Yv, float out_scale) {
  constexpr int K = EMBED;
  constexpr int N = EMBED;
  constexpr int XSTR = 40;  // 128 x (32 + pad) bf16
  constexpr int WSTR = 40;  //  64 x (32 + pad) bf16
  __shared__ __align__(16) __bf16 Xs[128 * XSTR];
  __shared__ __align__(16) __bf16 Ws[64 * WSTR];

  const int tid  = threadIdx.x;
  const int lane = tid & 31;
  const int wave = tid >> 5;
  const int grp  = (lane >> 4) & 1;  // half-wave group
  const int l16  = lane & 15;
  const int wm   = wave >> 1;        // 0..3 -> 32 rows each
  const int wn   = wave & 1;         // 0..1 -> 32 cols each

  const int nbase = blockIdx.x * 64;
  const int mbase = blockIdx.y * 128;

  v8f acc[2][2];
#pragma unroll
  for (int a = 0; a < 2; ++a)
#pragma unroll
    for (int b = 0; b < 2; ++b) acc[a][b] = vzero8();

  const float*  Xf = (const float*)Xv;
  const __bf16* Xb = (const __bf16*)Xv;

  for (int kb = 0; kb < K; kb += 32) {
    // ---- stage X tile: 128 rows x 32 cols, 2 threads/row x 16 cols ----
    {
      const int row = tid >> 1;
      const int c   = (tid & 1) * 16;
      if (XF32) {
        const float* src = Xf + (size_t)(mbase + row) * K + kb + c;
        v8bf lo, hi;
#pragma unroll
        for (int i = 0; i < 8; ++i) {
          lo[i] = (__bf16)src[i];
          hi[i] = (__bf16)src[8 + i];
        }
        *(v8bf*)&Xs[row * XSTR + c]     = lo;
        *(v8bf*)&Xs[row * XSTR + c + 8] = hi;
      } else {
        // bf16 source: copy memory->LDS on the async engine, no VGPR bounce
        async_copy_32B(Xb + (size_t)(mbase + row) * K + kb + c,
                       &Xs[row * XSTR + c]);
      }
    }
    // ---- stage W tile: 64 rows x 32 cols, 4 threads/row x 8 cols ----
    {
      const int row = tid >> 2;
      const int c   = (tid & 3) * 8;
      const float* src = W + (size_t)(nbase + row) * K + kb + c;
      v8bf v;
#pragma unroll
      for (int i = 0; i < 8; ++i) v[i] = (__bf16)src[i];
      *(v8bf*)&Ws[row * WSTR + c] = v;
    }
    if (!XF32) wait_async0();
    __syncthreads();

    // ---- fragments per ISA 7.12.2 layouts ----
    v16bf afrag[2], bfrag[2];
#pragma unroll
    for (int mt = 0; mt < 2; ++mt) {
      const int r   = wm * 32 + mt * 16 + l16;
      const int off = grp ? 8 : 0;  // A: K lo8 at grp*8, hi8 at 16+grp*8
      v8bf lo = *(const v8bf*)&Xs[r * XSTR + off];
      v8bf hi = *(const v8bf*)&Xs[r * XSTR + off + 16];
      afrag[mt] = cat8(lo, hi);
    }
#pragma unroll
    for (int nt = 0; nt < 2; ++nt) {
      const int r   = wn * 32 + nt * 16 + l16;
      const int off = grp ? 8 : 0;  // B: K lo8 at grp*8, hi8 at 16+grp*8
      v8bf lo = *(const v8bf*)&Ws[r * WSTR + off];
      v8bf hi = *(const v8bf*)&Ws[r * WSTR + off + 16];
      bfrag[nt] = cat8(lo, hi);
    }
#pragma unroll
    for (int mt = 0; mt < 2; ++mt)
#pragma unroll
      for (int nt = 0; nt < 2; ++nt)
        acc[mt][nt] = wmma_bf16(afrag[mt], bfrag[nt], acc[mt][nt]);

    __syncthreads();
  }

  // ---- epilogue: bias + scale + store ----
  float bn[2];
#pragma unroll
  for (int nt = 0; nt < 2; ++nt) bn[nt] = bias[nbase + wn * 32 + nt * 16 + l16];

#pragma unroll
  for (int mt = 0; mt < 2; ++mt) {
#pragma unroll
    for (int nt = 0; nt < 2; ++nt) {
#pragma unroll
      for (int i = 0; i < 8; ++i) {
        const int row = mbase + wm * 32 + mt * 16 + i + (grp ? 8 : 0);
        const int col = nbase + wn * 32 + nt * 16 + l16;
        const float v = (acc[mt][nt][i] + bn[nt]) * out_scale;
        if (OUTF32) {
          ((float*)Yv)[(size_t)row * N + col] = v;
        } else if (SPLITH) {
          const int b = row / TSEQ, t = row % TSEQ;
          const int h = col >> 6, d = col & 63;
          ((__bf16*)Yv)[(((size_t)b * NHEAD + h) * TSEQ + t) * HDIM + d] =
              (__bf16)v;
        } else {
          ((__bf16*)Yv)[(size_t)row * N + col] = (__bf16)v;
        }
      }
    }
  }
}

// ---------------------------------------------------------------------------
// Flash attention: one block = (b,h) x 64 query rows; 4 waves, each owns a
// 16-row Q tile. 64-key chunks: K staged via async-to-LDS DMA, V transposed
// into LDS through VGPRs. Row-sums of P via WMMA against a ones B-matrix;
// only the row-max needs cross-lane shuffles (4 steps per chunk per row).
// Q already carries the 1/sqrt(Dh) scale from its projection.
// ---------------------------------------------------------------------------
__global__ __launch_bounds__(128) void attention_kernel(
    const __bf16* __restrict__ qh, const __bf16* __restrict__ kh,
    const __bf16* __restrict__ vh, __bf16* __restrict__ out) {
  constexpr int CH   = 64;  // keys per chunk
  constexpr int KSTR = 72;  // 64 keys x (64 d + pad)
  constexpr int VTS  = 72;  // 64 d x (64 keys + pad), transposed V
  constexpr int PSTR = 64;  // 16 q x 64 keys per wave
  __shared__ __align__(16) __bf16 Ks[CH * KSTR];
  __shared__ __align__(16) __bf16 Vt[HDIM * VTS];
  __shared__ __align__(16) __bf16 Ps[4][16 * PSTR];

  const int tid  = threadIdx.x;
  const int lane = tid & 31;
  const int wave = tid >> 5;
  const int grp  = (lane >> 4) & 1;
  const int l16  = lane & 15;

  const int bh = blockIdx.y;  // 0..31 = b*16+h
  const int b  = bh >> 4;
  const int h  = bh & 15;
  const size_t head_base = (size_t)bh * TSEQ * HDIM;
  const int q0 = blockIdx.x * 64 + wave * 16;

  // Q A-fragments for d-chunks [0,32) and [32,64)
  v16bf qf[2];
  {
    const __bf16* qrow = qh + head_base + (size_t)(q0 + l16) * HDIM;
#pragma unroll
    for (int f = 0; f < 2; ++f) {
      const int off = f * 32 + (grp ? 8 : 0);
      v8bf lo = *(const v8bf*)(qrow + off);
      v8bf hi = *(const v8bf*)(qrow + off + 16);
      qf[f] = cat8(lo, hi);
    }
  }
  // ones B-fragment: row-sum of P as WMMA (P @ 1 -> row sums in all lanes)
  v16bf ones;
#pragma unroll
  for (int i = 0; i < 16; ++i) ones[i] = (__bf16)1.0f;

  v8f o[4];
#pragma unroll
  for (int t = 0; t < 4; ++t) o[t] = vzero8();
  float mrow[8], lrow[8];
#pragma unroll
  for (int i = 0; i < 8; ++i) {
    mrow[i] = -__builtin_inff();
    lrow[i] = 0.0f;
  }

  const int ldrow = tid >> 1;        // 0..63 local key / V row
  const int ldc   = (tid & 1) * 32;  // 0 or 32 d-offset

  for (int j0 = 0; j0 < TSEQ; j0 += CH) {
    __syncthreads();
    // ---- stage K (async DMA to LDS) and V (transposed via VGPRs) ----
    {
      async_copy_64B(kh + head_base + (size_t)(j0 + ldrow) * HDIM + ldc,
                     &Ks[ldrow * KSTR + ldc]);
      const __bf16* vsrc = vh + head_base + (size_t)(j0 + ldrow) * HDIM + ldc;
#pragma unroll
      for (int q = 0; q < 4; ++q) {
        v8bf vv = *(const v8bf*)(vsrc + q * 8);
#pragma unroll
        for (int i = 0; i < 8; ++i)
          Vt[(ldc + q * 8 + i) * VTS + ldrow] = vv[i];
      }
      if (j0 + CH < TSEQ) {  // hint next chunk toward the caches
        __builtin_prefetch(
            vh + head_base + (size_t)(j0 + CH + ldrow) * HDIM + ldc, 0, 1);
      }
    }
    wait_async0();
    __syncthreads();

    // ---- S = Q K^T for this 64-key chunk (four 16-key C tiles) ----
    v8f s[4];
#pragma unroll
    for (int jt = 0; jt < 4; ++jt) {
      s[jt] = vzero8();
      const int key = jt * 16 + l16;
#pragma unroll
      for (int dc = 0; dc < 2; ++dc) {
        const int doff = dc * 32 + (grp ? 8 : 0);
        v8bf lo = *(const v8bf*)&Ks[key * KSTR + doff];
        v8bf hi = *(const v8bf*)&Ks[key * KSTR + doff + 16];
        s[jt] = wmma_bf16(qf[dc], cat8(lo, hi), s[jt]);
      }
    }

    // ---- online softmax: row-max via 4 shuffle steps per row slot ----
    float alpha[8];
#pragma unroll
    for (int i = 0; i < 8; ++i) {
      float cm = fmaxf(fmaxf(s[0][i], s[1][i]), fmaxf(s[2][i], s[3][i]));
      cm = fmaxf(cm, __shfl_xor(cm, 1));
      cm = fmaxf(cm, __shfl_xor(cm, 2));
      cm = fmaxf(cm, __shfl_xor(cm, 4));
      cm = fmaxf(cm, __shfl_xor(cm, 8));
      const float mnew = fmaxf(mrow[i], cm);
      alpha[i] = __expf(mrow[i] - mnew);
      mrow[i] = mnew;
      // P in bf16 straight to LDS (C-layout -> re-read in A-layout)
      const int prow = i + (grp ? 8 : 0);
#pragma unroll
      for (int jt = 0; jt < 4; ++jt)
        Ps[wave][prow * PSTR + jt * 16 + l16] = (__bf16)__expf(s[jt][i] - mnew);
    }
#pragma unroll
    for (int t = 0; t < 4; ++t)
#pragma unroll
      for (int i = 0; i < 8; ++i) o[t][i] *= alpha[i];
#pragma unroll
    for (int i = 0; i < 8; ++i) lrow[i] *= alpha[i];

    // wave-private LDS bounce: ensure DS writes land before cross-lane reads
    asm volatile("s_wait_dscnt 0x0" ::: "memory");
    __builtin_amdgcn_wave_barrier();

    // ---- P A-fragments (16 q x 64 keys as two 16x32 frags) ----
    v16bf pf[2];
#pragma unroll
    for (int c2 = 0; c2 < 2; ++c2) {
      const int off = c2 * 32 + (grp ? 8 : 0);
      const __bf16* prow = &Ps[wave][l16 * PSTR];
      v8bf lo = *(const v8bf*)(prow + off);
      v8bf hi = *(const v8bf*)(prow + off + 16);
      pf[c2] = cat8(lo, hi);
    }
    // ---- row sums of P with WMMA: lsum[i] = sum_k P[row_i, k] ----
    {
      v8f lsum = wmma_bf16(pf[0], ones, vzero8());
      lsum = wmma_bf16(pf[1], ones, lsum);
#pragma unroll
      for (int i = 0; i < 8; ++i) lrow[i] += lsum[i];
    }
    // ---- O += P * V  (4 N-tiles of 16 dims x 2 key sub-chunks) ----
#pragma unroll
    for (int t = 0; t < 4; ++t) {
      const int d = t * 16 + l16;
#pragma unroll
      for (int c2 = 0; c2 < 2; ++c2) {
        const int off = c2 * 32 + (grp ? 8 : 0);
        v8bf lo = *(const v8bf*)&Vt[d * VTS + off];
        v8bf hi = *(const v8bf*)&Vt[d * VTS + off + 16];
        o[t] = wmma_bf16(pf[c2], cat8(lo, hi), o[t]);
      }
    }
  }

  // ---- epilogue: normalize and write merged heads [B*T, C] bf16 ----
#pragma unroll
  for (int i = 0; i < 8; ++i) {
    const float inv = 1.0f / lrow[i];
    const int row = q0 + i + (grp ? 8 : 0);
    const size_t obase = ((size_t)b * TSEQ + row) * EMBED + (size_t)h * HDIM;
#pragma unroll
    for (int t = 0; t < 4; ++t)
      out[obase + t * 16 + l16] = (__bf16)(o[t][i] * inv);
  }
}

// ---------------------------------------------------------------------------
extern "C" void kernel_launch(void* const* d_in, const int* in_sizes, int n_in,
                              void* d_out, int out_size, void* d_ws,
                              size_t ws_size, hipStream_t stream) {
  (void)in_sizes; (void)n_in; (void)out_size; (void)ws_size;
  const float* q  = (const float*)d_in[0];
  const float* k  = (const float*)d_in[1];
  const float* v  = (const float*)d_in[2];
  const float* Wq = (const float*)d_in[3];
  const float* bq = (const float*)d_in[4];
  const float* Wk = (const float*)d_in[5];
  const float* bk = (const float*)d_in[6];
  const float* Wv = (const float*)d_in[7];
  const float* bv = (const float*)d_in[8];
  const float* Wo = (const float*)d_in[9];
  const float* bo = (const float*)d_in[10];

  char* ws = (char*)d_ws;
  __bf16* qh = (__bf16*)(ws);                          //  8 MB [B,H,T,Dh]
  __bf16* kh = (__bf16*)(ws + ((size_t)8 << 20));      //  8 MB
  __bf16* vh = (__bf16*)(ws + ((size_t)16 << 20));     //  8 MB
  __bf16* ao = (__bf16*)(ws + ((size_t)24 << 20));     //  8 MB [B*T, C]
  float* out = (float*)d_out;

  const float qscale = 0.125f;  // 1/sqrt(HDIM), folded into Q projection

  dim3 gg(EMBED / 64, MROWS / 128);  // 16 x 32 blocks, 256 thr
  gemm_bias<true, true, false><<<gg, 256, 0, stream>>>(q, Wq, bq, qh, qscale);
  gemm_bias<true, true, false><<<gg, 256, 0, stream>>>(k, Wk, bk, kh, 1.0f);
  gemm_bias<true, true, false><<<gg, 256, 0, stream>>>(v, Wv, bv, vh, 1.0f);

  dim3 ga(TSEQ / 64, BATCH * NHEAD);  // 32 x 32 blocks, 128 thr
  attention_kernel<<<ga, 128, 0, stream>>>(qh, kh, vh, ao);

  gemm_bias<false, false, true><<<gg, 256, 0, stream>>>(ao, Wo, bo, (void*)out,
                                                        1.0f);
}
